// TransformerLayer_29128468201744
// MI455X (gfx1250) — compile-verified, matrix operand order checked by reading, not run on hardware
//
#include <hip/hip_runtime.h>
#include <hip/hip_bf16.h>
#include <stdint.h>

// GTrXL layer for MI455X (gfx1250): all matmuls via v_wmma_f32_16x16x32_f16,
// async global->LDS staging (ASYNCcnt, double-buffered), ds_load_b128 fragment
// loads, 32x32 C patch per wave (4 WMMAs / 8 LDS loads per K-step),
// f32 accumulation, fused bias/relu/gelu epilogues.

typedef __attribute__((ext_vector_type(16))) _Float16 v16h;
typedef __attribute__((ext_vector_type(8)))  _Float16 v8h;
typedef __attribute__((ext_vector_type(8)))  float    v8f;

#define BM 128
#define BN 64
#define BK 32
#define TPB 256
#define LDP (BK + 8)   // LDS row pitch in halves: 40 halves = 80 B (16B aligned)

union H4 { uint64_t u; _Float16 h[4]; };

__device__ __forceinline__ void async_b128(uint32_t lds_off, const _Float16* gaddr) {
  // CDNA5 GLOBAL_LOAD_ASYNC_TO_LDS_B128: per-lane 16B global -> LDS, tracked by ASYNCcnt.
  asm volatile("global_load_async_to_lds_b128 %0, %1, off"
               :: "v"(lds_off), "v"((unsigned long long)(uintptr_t)gaddr)
               : "memory");
}
__device__ __forceinline__ void wait_async0() {
  asm volatile("s_wait_asynccnt 0" ::: "memory");
}

__device__ __forceinline__ float act_apply(float x, int act) {
  if (act == 1) return x > 0.f ? x : 0.f;                       // relu
  if (act == 2) {                                               // tanh-gelu
    float x3 = x * x * x;
    return 0.5f * x * (1.f + tanhf(0.7978845608028654f * (x + 0.044715f * x3)));
  }
  return x;
}

// C[M,N] (f32 and/or f16) = act( A[M,K] * op(B) + bias ), batched over blockIdx.z.
// z -> (zb, zh) = (z / Hn, z % Hn); per-operand offsets zb*s?b + zh*s?h.
// transB==1: B is [N,K] row-major (C = A*B^T)  -> fully async contiguous staging.
// transB==0: B is [K,N] row-major              -> LDS-transpose staging.
// Requires M%128==0, N%64==0, K%32==0, 16B-aligned bases, 8-half-aligned strides.
__global__ __launch_bounds__(TPB) void wmma_gemm_kernel(
    const _Float16* __restrict__ A, long long lda, long long sAb, long long sAh,
    const _Float16* __restrict__ B, long long ldb, long long sBb, long long sBh,
    float* __restrict__ C, _Float16* __restrict__ Ch,
    long long ldc, long long sCb, long long sCh,
    const float* __restrict__ bias,
    int K, int Hn, int transB, int act)
{
  __shared__ __align__(16) _Float16 As[2][BM][LDP];
  __shared__ __align__(16) _Float16 Bt[2][BN][LDP];   // stored as [n][k]

  const int tid  = threadIdx.x;
  const int lane = tid & 31;
  const int wave = tid >> 5;
  const int wm   = (wave & 3) << 5;   // wave row in tile: 0,32,64,96
  const int wn   = (wave >> 2) << 5;  // wave col in tile: 0,32
  const int l16  = lane & 15;
  const int hsel = lane >> 4;         // K/M half owned by this lane

  const int zb = blockIdx.z / Hn;
  const int zh = blockIdx.z % Hn;
  const long long bn0 = (long long)blockIdx.x * BN;
  const long long bm0 = (long long)blockIdx.y * BM;

  const _Float16* Ab = A + (long long)zb * sAb + (long long)zh * sAh + bm0 * lda;
  const _Float16* Bb = B + (long long)zb * sBb + (long long)zh * sBh;

  auto stage = [&](int k0, int buf) {
    // A: 128x32 halves = 512 b128 chunks -> 2 per thread
#pragma unroll
    for (int t = 0; t < 2; ++t) {
      int cid = tid + t * TPB;
      int r = cid >> 2;               // 0..127
      int c = (cid & 3) << 3;         // 0,8,16,24
      async_b128((uint32_t)(uintptr_t)&As[buf][r][c], Ab + (long long)r * lda + k0 + c);
    }
    if (transB) {
      int n = tid >> 2;               // 0..63
      int c = (tid & 3) << 3;
      async_b128((uint32_t)(uintptr_t)&Bt[buf][n][c], Bb + (bn0 + n) * ldb + k0 + c);
    } else {
      // B[K,N]: read rows (contiguous in n), transpose into Bt[n][k]
#pragma unroll
      for (int t = 0; t < 2; ++t) {
        int i = tid * 2 + t;
        int r = i >> 4;               // k within panel: 0..31
        int c = (i & 15) << 2;        // n chunk: 0..60
        H4 d; d.u = *(const uint64_t*)(Bb + (long long)(k0 + r) * ldb + bn0 + c);
        Bt[buf][c + 0][r] = d.h[0];
        Bt[buf][c + 1][r] = d.h[1];
        Bt[buf][c + 2][r] = d.h[2];
        Bt[buf][c + 3][r] = d.h[3];
      }
    }
  };

  v8f acc00 = {}, acc01 = {}, acc10 = {}, acc11 = {};

  stage(0, 0);
  wait_async0();
  __syncthreads();

  int cur = 0;
  for (int k0 = 0; k0 < K; k0 += BK) {
    if (k0 + BK < K) stage(k0 + BK, cur ^ 1);   // prefetch next panel (double buffer)

    // Fragment loads: two contiguous 16B chunks per matrix per lane (ds_load_b128).
    // v16h element order == {K hsel*8..+7, K 16+hsel*8..+7} == chunk[hsel], chunk[2+hsel].
    const v8h* pa0 = (const v8h*)&As[cur][wm + l16][0];
    const v8h* pa1 = (const v8h*)&As[cur][wm + 16 + l16][0];
    const v8h* pb0 = (const v8h*)&Bt[cur][wn + l16][0];
    const v8h* pb1 = (const v8h*)&Bt[cur][wn + 16 + l16][0];
    v16h af0 = __builtin_shufflevector(pa0[hsel], pa0[2 + hsel],
                                       0,1,2,3,4,5,6,7,8,9,10,11,12,13,14,15);
    v16h af1 = __builtin_shufflevector(pa1[hsel], pa1[2 + hsel],
                                       0,1,2,3,4,5,6,7,8,9,10,11,12,13,14,15);
    v16h bf0 = __builtin_shufflevector(pb0[hsel], pb0[2 + hsel],
                                       0,1,2,3,4,5,6,7,8,9,10,11,12,13,14,15);
    v16h bf1 = __builtin_shufflevector(pb1[hsel], pb1[2 + hsel],
                                       0,1,2,3,4,5,6,7,8,9,10,11,12,13,14,15);

    acc00 = __builtin_amdgcn_wmma_f32_16x16x32_f16(false, af0, false, bf0, (short)0, acc00, false, false);
    acc01 = __builtin_amdgcn_wmma_f32_16x16x32_f16(false, af0, false, bf1, (short)0, acc01, false, false);
    acc10 = __builtin_amdgcn_wmma_f32_16x16x32_f16(false, af1, false, bf0, (short)0, acc10, false, false);
    acc11 = __builtin_amdgcn_wmma_f32_16x16x32_f16(false, af1, false, bf1, (short)0, acc11, false, false);

    wait_async0();          // next panel landed (own wave)
    __syncthreads();        // all waves done reading cur / writing next
    cur ^= 1;
  }

  // Epilogue: C layout VGPR v -> M = v + 8*hsel, N = lane&15
  float*    Cb  = C  ? (C  + (long long)zb * sCb + (long long)zh * sCh) : nullptr;
  _Float16* Chb = Ch ? (Ch + (long long)zb * sCb + (long long)zh * sCh) : nullptr;
  const long long gn0 = bn0 + wn + l16;
  const long long gn1 = gn0 + 16;
  float bs0 = bias ? bias[gn0] : 0.f;
  float bs1 = bias ? bias[gn1] : 0.f;
#pragma unroll
  for (int v = 0; v < 8; ++v) {
    long long gm0 = bm0 + wm + v + (hsel << 3);
    long long gm1 = gm0 + 16;
    float r00 = act_apply(acc00[v] + bs0, act);
    float r01 = act_apply(acc01[v] + bs1, act);
    float r10 = act_apply(acc10[v] + bs0, act);
    float r11 = act_apply(acc11[v] + bs1, act);
    if (Cb) {
      Cb[gm0 * ldc + gn0] = r00;  Cb[gm0 * ldc + gn1] = r01;
      Cb[gm1 * ldc + gn0] = r10;  Cb[gm1 * ldc + gn1] = r11;
    }
    if (Chb) {
      Chb[gm0 * ldc + gn0] = (_Float16)r00;  Chb[gm0 * ldc + gn1] = (_Float16)r01;
      Chb[gm1 * ldc + gn0] = (_Float16)r10;  Chb[gm1 * ldc + gn1] = (_Float16)r11;
    }
  }
}

// Row LayerNorm (D reduced per block), f32 in -> f16 out.
__global__ __launch_bounds__(TPB) void ln_kernel(const float* __restrict__ x,
                                                 const float* __restrict__ s,
                                                 const float* __restrict__ b,
                                                 _Float16* __restrict__ out, int D)
{
  __shared__ float red[TPB];
  const int tid = threadIdx.x;
  const long long row = blockIdx.x;
  const float* xr = x + row * (long long)D;

  float ls = 0.f;
  for (int c = tid; c < D; c += TPB) ls += xr[c];
  red[tid] = ls; __syncthreads();
  for (int st = TPB / 2; st > 0; st >>= 1) { if (tid < st) red[tid] += red[tid + st]; __syncthreads(); }
  float mu = red[0] / (float)D;
  __syncthreads();

  float lv = 0.f;
  for (int c = tid; c < D; c += TPB) { float dd = xr[c] - mu; lv += dd * dd; }
  red[tid] = lv; __syncthreads();
  for (int st = TPB / 2; st > 0; st >>= 1) { if (tid < st) red[tid] += red[tid + st]; __syncthreads(); }
  float inv = rsqrtf(red[0] / (float)D + 1e-6f);

  _Float16* orow = out + row * (long long)D;
  for (int c = tid; c < D; c += TPB)
    orow[c] = (_Float16)((xr[c] - mu) * inv * s[c] + b[c]);
}

__global__ void cvt_f32_f16_kernel(const float* __restrict__ in, _Float16* __restrict__ out, long long n)
{
  long long i = (long long)blockIdx.x * blockDim.x + threadIdx.x;
  long long stride = (long long)gridDim.x * blockDim.x;
  for (; i < n; i += stride) out[i] = (_Float16)in[i];
}

// f32 [K,N] -> f16 [N,K] transpose (32x32 LDS tile); K,N multiples of 32.
__global__ __launch_bounds__(1024) void cvt_transpose_kernel(const float* __restrict__ in,
                                                             _Float16* __restrict__ out,
                                                             int K, int N)
{
  __shared__ _Float16 tile[32][33];
  const int tx = threadIdx.x & 31;
  const int ty = threadIdx.x >> 5;
  const long long nb = (long long)blockIdx.x * 32;
  const long long kb = (long long)blockIdx.y * 32;
  tile[ty][tx] = (_Float16)in[(kb + ty) * N + nb + tx];
  __syncthreads();
  out[(nb + ty) * K + kb + tx] = tile[tx][ty];
}

// out[i] = in[i] + vec[i % D]   (f16 + f32 -> f16); used for q + r_r_bias
__global__ void add_vec_kernel(const _Float16* __restrict__ in, const float* __restrict__ vec,
                               _Float16* __restrict__ out, int D, long long n)
{
  long long i = (long long)blockIdx.x * blockDim.x + threadIdx.x;
  if (i >= n) return;
  out[i] = (_Float16)((float)in[i] + vec[(int)(i % D)]);
}

// cb[b,h,k] = scale * dot(r_w_bias[h,:], k[b,k,h,:])
__global__ void content_bias_kernel(const _Float16* __restrict__ kh, const float* __restrict__ rwb,
                                    float* __restrict__ cb, int Bn, int S, int H, int HD, int D,
                                    float scale)
{
  long long t = (long long)blockIdx.x * blockDim.x + threadIdx.x;
  long long total = (long long)Bn * H * S;
  if (t >= total) return;
  int k = (int)(t % S);
  int h = (int)((t / S) % H);
  int b = (int)(t / ((long long)S * H));
  const _Float16* kp = kh + ((long long)(b * S + k)) * D + h * HD;
  const float* wb = rwb + h * HD;
  float acc = 0.f;
  for (int d = 0; d < HD; ++d) acc += wb[d] * (float)kp[d];
  cb[t] = acc * scale;
}

// Fused: scores = qk*scale + content_bias + rel_shift(BD)*scale + causal mask; softmax -> f16.
// One block per (b,h,q) row; assumes S == 4*TPB.
__global__ __launch_bounds__(TPB) void softmax_kernel(
    const float* __restrict__ qk, const _Float16* __restrict__ bd,
    const float* __restrict__ cb, _Float16* __restrict__ probs,
    int S, int H, float scale)
{
  __shared__ float red[TPB];
  const int tid = threadIdx.x;
  long long bz = blockIdx.x;
  int q = (int)(bz % S);
  int h = (int)((bz / S) % H);
  int b = (int)(bz / ((long long)S * H));
  long long row = ((long long)(b * H + h) * S + q) * (long long)S;
  const float*    qkr = qk + row;
  const _Float16* bdr = bd + row;
  const float*    cbr = cb + (long long)(b * H + h) * S;
  _Float16*       pr  = probs + row;

  float vals[4];
  float m = -3.0e38f;
#pragma unroll
  for (int u = 0; u < 4; ++u) {
    int k = tid + u * TPB;
    float vv = -3.0e38f;
    if (k <= q) {                       // causal; rel_shift: out[q,k] = BD[q, k-q+S-1]
      int j = k - q + S - 1;
      vv = qkr[k] * scale + cbr[k] + (float)bdr[j] * scale;
    }
    vals[u] = vv;
    m = fmaxf(m, vv);
  }
  red[tid] = m; __syncthreads();
  for (int st = TPB / 2; st > 0; st >>= 1) { if (tid < st) red[tid] = fmaxf(red[tid], red[tid + st]); __syncthreads(); }
  m = red[0]; __syncthreads();

  float sm = 0.f;
#pragma unroll
  for (int u = 0; u < 4; ++u) {
    vals[u] = (vals[u] <= -1.0e38f) ? 0.f : expf(vals[u] - m);
    sm += vals[u];
  }
  red[tid] = sm; __syncthreads();
  for (int st = TPB / 2; st > 0; st >>= 1) { if (tid < st) red[tid] += red[tid + st]; __syncthreads(); }
  float inv = 1.f / red[0];
#pragma unroll
  for (int u = 0; u < 4; ++u) pr[tid + u * TPB] = (_Float16)(vals[u] * inv);
}

// GRU gate part 1: r,z from xrz/yrzh; emit rx=(r*x) as f16 and z as f32.
__global__ void gate_rz_kernel(const float* __restrict__ xrz, const float* __restrict__ yrzh,
                               const float* __restrict__ x, const float* __restrict__ gb,
                               _Float16* __restrict__ rx, float* __restrict__ zb,
                               int D, long long n)
{
  long long i = (long long)blockIdx.x * blockDim.x + threadIdx.x;
  if (i >= n) return;
  long long row = i / D;
  int d = (int)(i % D);
  float xr = xrz[row * (2ll * D) + d];
  float xz = xrz[row * (2ll * D) + D + d];
  float yr = yrzh[row * (3ll * D) + d];
  float yz = yrzh[row * (3ll * D) + D + d];
  float r = 1.f / (1.f + expf(-(yr + xr)));
  float z = 1.f / (1.f + expf(-(yz + xz - gb[d])));
  rx[i] = (_Float16)(r * x[i]);
  zb[i] = z;
}

// GRU gate part 2: out = (1-z)*x + z*tanh(y_h + (r*x)@W_xh)
__global__ void gate_out_kernel(const float* __restrict__ yrzh, const float* __restrict__ hh,
                                const float* __restrict__ x, const float* __restrict__ zb,
                                float* __restrict__ out, _Float16* __restrict__ outh,
                                int D, long long n)
{
  long long i = (long long)blockIdx.x * blockDim.x + threadIdx.x;
  if (i >= n) return;
  long long row = i / D;
  int d = (int)(i % D);
  float yh = yrzh[row * (3ll * D) + 2ll * D + d];
  float hv = tanhf(yh + hh[i]);
  float z  = zb[i];
  float o  = (1.f - z) * x[i] + z * hv;
  out[i] = o;
  if (outh) outh[i] = (_Float16)o;
}

extern "C" void kernel_launch(void* const* d_in, const int* in_sizes, int n_in,
                              void* d_out, int out_size, void* d_ws, size_t ws_size,
                              hipStream_t stream)
{
  (void)in_sizes; (void)n_in; (void)out_size; (void)ws_size;
  const int Bb = 4, S = 1024, D = 1024, H = 16, HD = 64, MLPD = 4096;
  const long long NT = (long long)Bb * S;           // 4096 tokens
  const float scale = 0.125f;                       // HD^-0.5

  // ---- inputs (setup_inputs order) ----
  const float* values_keys = (const float*)d_in[0];
  const float* queries     = (const float*)d_in[1];
  const float* ln1_s = (const float*)d_in[4];
  const float* ln1_b = (const float*)d_in[5];
  const float* ln2_s = (const float*)d_in[6];
  const float* ln2_b = (const float*)d_in[7];
  const float* bq = (const float*)d_in[9];
  const float* bk = (const float*)d_in[11];
  const float* bv = (const float*)d_in[13];
  const float* r_w_bias = (const float*)d_in[15];
  const float* r_r_bias = (const float*)d_in[16];
  const float* bo = (const float*)d_in[18];
  const float* b1 = (const float*)d_in[20];
  const float* b2 = (const float*)d_in[22];
  const float* g1_b = (const float*)d_in[26];
  const float* g2_b = (const float*)d_in[30];

  // ---- workspace bump allocator ----
  char* wp = (char*)d_ws;
  auto alloc = [&](size_t bytes) -> void* {
    void* r = (void*)wp;
    wp += (bytes + 255) & ~(size_t)255;
    return r;
  };
  const size_t DD = (size_t)D * D;

  // transposed f16 weights [N,K]
  _Float16* wWq   = (_Float16*)alloc(DD * 2);
  _Float16* wWk   = (_Float16*)alloc(DD * 2);
  _Float16* wWv   = (_Float16*)alloc(DD * 2);
  _Float16* wWr   = (_Float16*)alloc(DD * 2);
  _Float16* wWo   = (_Float16*)alloc(DD * 2);
  _Float16* wW1   = (_Float16*)alloc((size_t)D * MLPD * 2);
  _Float16* wW2   = (_Float16*)alloc((size_t)MLPD * D * 2);
  _Float16* w1xrz = (_Float16*)alloc(DD * 2 * 2);
  _Float16* w1yrzh= (_Float16*)alloc(DD * 3 * 2);
  _Float16* w1xh  = (_Float16*)alloc(DD * 2);
  _Float16* w2xrz = (_Float16*)alloc(DD * 2 * 2);
  _Float16* w2yrzh= (_Float16*)alloc(DD * 3 * 2);
  _Float16* w2xh  = (_Float16*)alloc(DD * 2);
  _Float16* peh   = (_Float16*)alloc((size_t)S * D * 2);

  _Float16* kvnh  = (_Float16*)alloc((size_t)NT * D * 2);
  _Float16* qnh   = (_Float16*)alloc((size_t)NT * D * 2);
  _Float16* qh    = (_Float16*)alloc((size_t)NT * D * 2);
  _Float16* qrrh  = (_Float16*)alloc((size_t)NT * D * 2);
  _Float16* kh    = (_Float16*)alloc((size_t)NT * D * 2);
  _Float16* vh    = (_Float16*)alloc((size_t)NT * D * 2);
  _Float16* rh    = (_Float16*)alloc((size_t)S * D * 2);
  _Float16* attnh = (_Float16*)alloc((size_t)NT * D * 2);
  _Float16* y1h   = (_Float16*)alloc((size_t)NT * D * 2);
  _Float16* xqh   = (_Float16*)alloc((size_t)NT * D * 2);
  _Float16* rxh   = (_Float16*)alloc((size_t)NT * D * 2);
  _Float16* out1h = (_Float16*)alloc((size_t)NT * D * 2);
  _Float16* ln2h  = (_Float16*)alloc((size_t)NT * D * 2);
  _Float16* y2h   = (_Float16*)alloc((size_t)NT * D * 2);
  _Float16* mlph  = (_Float16*)alloc((size_t)NT * MLPD * 2);
  _Float16* probsh= (_Float16*)alloc((size_t)Bb * H * S * S * 2);
  _Float16* bdh   = (_Float16*)alloc((size_t)Bb * H * S * S * 2);

  float* qkf   = (float*)alloc((size_t)Bb * H * S * S * 4);
  float* cbf   = (float*)alloc((size_t)Bb * H * S * 4);
  float* xrzf  = (float*)alloc((size_t)NT * 2 * D * 4);
  float* yrzhf = (float*)alloc((size_t)NT * 3 * D * 4);
  float* zbuf  = (float*)alloc((size_t)NT * D * 4);
  float* hhf   = (float*)alloc((size_t)NT * D * 4);
  float* out1f = (float*)alloc((size_t)NT * D * 4);

  // ---- helpers ----
  auto cvt = [&](const void* src, _Float16* dst, long long n) {
    cvt_f32_f16_kernel<<<1024, TPB, 0, stream>>>((const float*)src, dst, n);
  };
  auto cvtT = [&](const void* src, _Float16* dst, int K, int N) {   // [K,N] -> [N,K]
    dim3 grid(N / 32, K / 32);
    cvt_transpose_kernel<<<grid, 1024, 0, stream>>>((const float*)src, dst, K, N);
  };
  auto gemm = [&](const _Float16* A, long long lda, long long sAb, long long sAh,
                  const _Float16* Bm, long long ldb, long long sBb, long long sBh,
                  float* C, _Float16* Ch, long long ldc, long long sCb, long long sCh,
                  const float* bias, int M, int Nn, int K, int Hn, int Z, int transB, int act) {
    dim3 grid(Nn / BN, M / BM, Z);
    wmma_gemm_kernel<<<grid, TPB, 0, stream>>>(A, lda, sAb, sAh, Bm, ldb, sBb, sBh,
                                               C, Ch, ldc, sCb, sCh, bias, K, Hn, transB, act);
  };
  const int EW = (int)((NT * D) / TPB);

  // ---- 0) convert (and transpose) weights; f16 copies of pos_embed / queries ----
  cvtT(d_in[8],  wWq,    D, D);
  cvtT(d_in[10], wWk,    D, D);
  cvtT(d_in[12], wWv,    D, D);
  cvtT(d_in[14], wWr,    D, D);
  cvtT(d_in[17], wWo,    D, D);
  cvtT(d_in[19], wW1,    D, MLPD);
  cvtT(d_in[21], wW2,    MLPD, D);
  cvtT(d_in[23], w1xrz,  D, 2 * D);
  cvtT(d_in[24], w1yrzh, D, 3 * D);
  cvtT(d_in[25], w1xh,   D, D);
  cvtT(d_in[27], w2xrz,  D, 2 * D);
  cvtT(d_in[28], w2yrzh, D, 3 * D);
  cvtT(d_in[29], w2xh,   D, D);
  cvt(d_in[2], peh, (long long)S * D);
  cvt(d_in[1], xqh, NT * D);

  // ---- 1) shared pre-LN ----
  ln_kernel<<<(int)NT, TPB, 0, stream>>>(values_keys, ln1_s, ln1_b, kvnh, D);
  ln_kernel<<<(int)NT, TPB, 0, stream>>>(queries,     ln1_s, ln1_b, qnh,  D);

  // ---- 2) projections (all NT against transposed weights) ----
  gemm(qnh,  D,0,0, wWq, D,0,0, nullptr, qh, D,0,0, bq, (int)NT, D, D, 1,1, 1,0);
  add_vec_kernel<<<EW, TPB, 0, stream>>>(qh, r_r_bias, qrrh, D, NT * D);
  gemm(kvnh, D,0,0, wWk, D,0,0, nullptr, kh, D,0,0, bk, (int)NT, D, D, 1,1, 1,0);
  gemm(kvnh, D,0,0, wWv, D,0,0, nullptr, vh, D,0,0, bv, (int)NT, D, D, 1,1, 1,0);
  gemm(peh,  D,0,0, wWr, D,0,0, nullptr, rh, D,0,0, nullptr, S, D, D, 1,1, 1,0);

  // ---- 3) attention ----
  content_bias_kernel<<<(Bb * H * S) / TPB, TPB, 0, stream>>>(kh, r_w_bias, cbf, Bb, S, H, HD, D, scale);
  // qk[b,h,q,k] = q . k^T  (NT, K=HD, batched over B*H)
  gemm(qh,   D, (long long)S * D, HD,  kh, D, (long long)S * D, HD,
       qkf, nullptr, S, (long long)H * S * S, (long long)S * S, nullptr, S, S, HD, H, Bb * H, 1, 0);
  // BD[b,h,q,j] = (q+r_r_bias) . r^T  (r shared across batch)
  gemm(qrrh, D, (long long)S * D, HD,  rh, D, 0, HD,
       nullptr, bdh, S, (long long)H * S * S, (long long)S * S, nullptr, S, S, HD, H, Bb * H, 1, 0);
  softmax_kernel<<<Bb * H * S, TPB, 0, stream>>>(qkf, bdh, cbf, probsh, S, H, scale);
  // attn[b,q,h,:] = probs . v  (NN, K=S, batched over B*H, N=HD)
  gemm(probsh, S, (long long)H * S * S, (long long)S * S,  vh, D, (long long)S * D, HD,
       nullptr, attnh, D, (long long)S * D, HD, nullptr, S, HD, S, H, Bb * H, 0, 0);
  // y1 = relu(attn @ Wo + bo)
  gemm(attnh, D,0,0, wWo, D,0,0, nullptr, y1h, D,0,0, bo, (int)NT, D, D, 1,1, 1,1);

  // ---- 4) gate 1: x = queries, y = y1 ----
  gemm(xqh, D,0,0, w1xrz,  D,0,0, xrzf,  nullptr, 2*D,0,0, nullptr, (int)NT, 2*D, D, 1,1, 1,0);
  gemm(y1h, D,0,0, w1yrzh, D,0,0, yrzhf, nullptr, 3*D,0,0, nullptr, (int)NT, 3*D, D, 1,1, 1,0);
  gate_rz_kernel<<<EW, TPB, 0, stream>>>(xrzf, yrzhf, queries, g1_b, rxh, zbuf, D, NT * D);
  gemm(rxh, D,0,0, w1xh, D,0,0, hhf, nullptr, D,0,0, nullptr, (int)NT, D, D, 1,1, 1,0);
  gate_out_kernel<<<EW, TPB, 0, stream>>>(yrzhf, hhf, queries, zbuf, out1f, out1h, D, NT * D);

  // ---- 5) MLP ----
  ln_kernel<<<(int)NT, TPB, 0, stream>>>(out1f, ln2_s, ln2_b, ln2h, D);
  gemm(ln2h, D,0,0,    wW1, D,0,0,    nullptr, mlph, MLPD,0,0, b1, (int)NT, MLPD, D,    1,1, 1,2); // gelu
  gemm(mlph, MLPD,0,0, wW2, MLPD,0,0, nullptr, y2h,  D,0,0,    b2, (int)NT, D,    MLPD, 1,1, 1,1); // relu

  // ---- 6) gate 2: x = out1, y = y2; final output to d_out ----
  gemm(out1h, D,0,0, w2xrz,  D,0,0, xrzf,  nullptr, 2*D,0,0, nullptr, (int)NT, 2*D, D, 1,1, 1,0);
  gemm(y2h,   D,0,0, w2yrzh, D,0,0, yrzhf, nullptr, 3*D,0,0, nullptr, (int)NT, 3*D, D, 1,1, 1,0);
  gate_rz_kernel<<<EW, TPB, 0, stream>>>(xrzf, yrzhf, out1f, g2_b, rxh, zbuf, D, NT * D);
  gemm(rxh, D,0,0, w2xh, D,0,0, hhf, nullptr, D,0,0, nullptr, (int)NT, D, D, 1,1, 1,0);
  gate_out_kernel<<<EW, TPB, 0, stream>>>(yrzhf, hhf, out1f, zbuf, (float*)d_out, nullptr, D, NT * D);
}
